// SparseLinear_6588479832125
// MI455X (gfx1250) — matrix operand order repaired and checked by reading, not compiled, not run on hardware
//
#include <hip/hip_runtime.h>
#include <stdint.h>

// Problem constants (match reference)
#define M_DIM 4096
#define K_DIM 4096
#define B_DIM 4
#define N_DIM 2048

// GEMM tiling
#define BM 128
#define BN 128
#define BK 32
#define NK_ITERS (K_DIM / BK)  // 128

typedef __attribute__((ext_vector_type(16))) __bf16 v16bf;
typedef __attribute__((ext_vector_type(8)))  float  v8f;

union Frag16 { v16bf f; uint4 q[2]; };

// Packed f32x2 -> bf16x2 (RNE) via the hardware converter.
// Mnemonic confirmed present on gfx1250 (emitted by the compiler in round 4);
// direct asm avoids the half-repacking shuffles the generic lowering adds.
__device__ __forceinline__ unsigned int pk_bf16(float a, float b) {
    unsigned int r;
    asm("v_cvt_pk_bf16_f32 %0, %1, %2" : "=v"(r) : "v"(a), "v"(b));
    return r;
}
__device__ __forceinline__ unsigned short f2bf(float a) {
    union { __bf16 h; unsigned short u; } cv;
    cv.h = (__bf16)a;
    return cv.u;
}

// CDNA5 async copy: global -> LDS, 32 bytes per lane (two b128 transfers).
// INST_OFFSET is added to BOTH the LDS and global addresses (ISA 08 §4.4),
// so a single (lds, gaddr) pair covers both chunks.
__device__ __forceinline__ void async_copy32(uint32_t lds_addr, uint64_t gaddr) {
    asm volatile("global_load_async_to_lds_b128 %0, %1, off"
                 :: "v"(lds_addr), "v"(gaddr) : "memory");
    asm volatile("global_load_async_to_lds_b128 %0, %1, off offset:16"
                 :: "v"(lds_addr), "v"(gaddr) : "memory");
}
__device__ __forceinline__ void wait_async0() {
    asm volatile("s_wait_asynccnt 0x0" ::: "memory");
}

// ---------------------------------------------------------------------------
// Kernel 1: scatter CSR row -> dense bf16 row of W (duplicate cols summed).
// ---------------------------------------------------------------------------
__global__ __launch_bounds__(256)
void scatter_w_kernel(const float* __restrict__ values,
                      const int*   __restrict__ row_offsets,
                      const int*   __restrict__ cols,
                      unsigned short* __restrict__ Wbf) {
    __shared__ float row[K_DIM];
    const int m = blockIdx.x;
    for (int i = threadIdx.x; i < K_DIM; i += blockDim.x) row[i] = 0.0f;
    __syncthreads();
    const int r0 = row_offsets[m];
    const int r1 = row_offsets[m + 1];
    for (int i = r0 + threadIdx.x; i < r1; i += blockDim.x) {
        atomicAdd(&row[cols[i]], values[i]);
    }
    __syncthreads();
    unsigned short* wrow = Wbf + (size_t)m * K_DIM;
    for (int i = threadIdx.x; i < K_DIM; i += blockDim.x) {
        wrow[i] = f2bf(row[i]);
    }
}

// ---------------------------------------------------------------------------
// Kernel 2: dense bf16 WMMA GEMM  out[b,m,n] = sum_k W[m,k] * x[b,n,k]
// 128x128 block tile, 8 wave32 (2 M x 4 N), wave tile 64x32 = 4x2 WMMA frags.
// Double-buffered LDS; A tile via async-to-LDS engine, B tile converted
// f32->bf16 with v_cvt_pk_bf16_f32.
// ---------------------------------------------------------------------------
__global__ __launch_bounds__(256)
void spmm_gemm_kernel(const float* __restrict__ x,
                      const unsigned short* __restrict__ Wbf,
                      float* __restrict__ out) {
    __shared__ __align__(16) unsigned short As[2][BM * BK];  // W tile  [m][k]
    __shared__ __align__(16) unsigned short Bs[2][BN * BK];  // x tile  [n][k]

    const int m0   = blockIdx.x * BM;
    const int n0   = blockIdx.y * BN;
    const int b    = blockIdx.z;
    const int t    = threadIdx.x;
    const int lane = t & 31;
    const int wave = t >> 5;
    const int wm   = (wave & 1) * 64;
    const int wn   = (wave >> 1) * 32;

    // Tile staging: 256 threads cover 128 rows x 2 k-segments of 16 elements
    const int lrow = t >> 1;
    const int kseg = (t & 1) * 16;

    const float* xb = x + (size_t)b * N_DIM * K_DIM;
    const unsigned short* aptr = Wbf + (size_t)(m0 + lrow) * K_DIM + kseg;
    const float*          bptr = xb  + (size_t)(n0 + lrow) * K_DIM + kseg;

    v8f acc[4][2] = {};

    // ---- prologue: async A tile 0 -> LDS[0]; B tile 0 -> registers ----
    async_copy32((uint32_t)(uintptr_t)&As[0][lrow * BK + kseg],
                 (uint64_t)(uintptr_t)aptr);
    float4 f0, f1, f2, f3;
    {
        const float4* gX = (const float4*)bptr;
        f0 = gX[0]; f1 = gX[1]; f2 = gX[2]; f3 = gX[3];
    }

    int cur = 0;
    for (int kt = 0; kt < NK_ITERS; ++kt) {
        // ---- store B regs (bf16) -> LDS_B[cur] ----
        uint4 p0, p1;
        p0.x = pk_bf16(f0.x, f0.y); p0.y = pk_bf16(f0.z, f0.w);
        p0.z = pk_bf16(f1.x, f1.y); p0.w = pk_bf16(f1.z, f1.w);
        p1.x = pk_bf16(f2.x, f2.y); p1.y = pk_bf16(f2.z, f2.w);
        p1.z = pk_bf16(f3.x, f3.y); p1.w = pk_bf16(f3.z, f3.w);
        *(uint4*)(&Bs[cur][lrow * BK + kseg])     = p0;
        *(uint4*)(&Bs[cur][lrow * BK + kseg + 8]) = p1;

        wait_async0();      // drain this wave's A-tile asyncs for tile kt
        __syncthreads();    // all A asyncs + B stores visible block-wide

        // ---- prefetch tile kt+1 (overlaps WMMA below) ----
        if (kt + 1 < NK_ITERS) {
            const int k1 = (kt + 1) * BK;
            async_copy32((uint32_t)(uintptr_t)&As[cur ^ 1][lrow * BK + kseg],
                         (uint64_t)(uintptr_t)(aptr + k1));
            const float4* gX = (const float4*)(bptr + k1);
            f0 = gX[0]; f1 = gX[1]; f2 = gX[2]; f3 = gX[3];
        }

        // ---- build fragments per ISA lane layouts ----
        // A (16x32 bf16): lane L -> row m=L%16; K = kb..kb+7, kb+16..kb+23,
        // kb = (L<16) ? 0 : 8.
        Frag16 af[4];
        {
            const int kb = (lane < 16) ? 0 : 8;
            const int mr = lane & 15;
#pragma unroll
            for (int i = 0; i < 4; ++i) {
                const int mrow = wm + 16 * i + mr;
                af[i].q[0] = *(const uint4*)(&As[cur][mrow * BK + kb]);
                af[i].q[1] = *(const uint4*)(&As[cur][mrow * BK + kb + 16]);
            }
        }
        // B (32x16 bf16): lane L -> column n=L%16; K = kb..kb+7, kb+16..kb+23,
        // kb = (L<16) ? 0 : 8.  B[k][n] = x[n][k].
        Frag16 bf[2];
        {
            const int kb = (lane < 16) ? 0 : 8;
            const int nr = lane & 15;
#pragma unroll
            for (int j = 0; j < 2; ++j) {
                const int nrow = wn + 16 * j + nr;
                bf[j].q[0] = *(const uint4*)(&Bs[cur][nrow * BK + kb]);
                bf[j].q[1] = *(const uint4*)(&Bs[cur][nrow * BK + kb + 16]);
            }
        }

        // ---- 8 WMMAs per wave per k-step ----
#pragma unroll
        for (int i = 0; i < 4; ++i) {
#pragma unroll
            for (int j = 0; j < 2; ++j) {
                acc[i][j] = __builtin_amdgcn_wmma_f32_16x16x32_bf16(
                    false, af[i].f, false, bf[j].f,
                    (short)0, acc[i][j], false, false);
            }
        }

        cur ^= 1;
    }

    // ---- epilogue: D layout (VGPR r, lane L: M=r+8*(L>=16), N=L%16) ----
    const int nn   = lane & 15;
    const int mofs = (lane >> 4) * 8;
    float* ob = out + (size_t)b * M_DIM * N_DIM;
#pragma unroll
    for (int i = 0; i < 4; ++i) {
#pragma unroll
        for (int j = 0; j < 2; ++j) {
#pragma unroll
            for (int r = 0; r < 8; ++r) {
                const int mrow = m0 + wm + 16 * i + mofs + r;
                const int ncol = n0 + wn + 16 * j + nn;
                ob[(size_t)mrow * N_DIM + ncol] = acc[i][j][r];
            }
        }
    }
}

// ---------------------------------------------------------------------------
extern "C" void kernel_launch(void* const* d_in, const int* in_sizes, int n_in,
                              void* d_out, int out_size, void* d_ws, size_t ws_size,
                              hipStream_t stream) {
    (void)in_sizes; (void)n_in; (void)out_size; (void)ws_size;
    const float* x           = (const float*)d_in[0];
    const float* values      = (const float*)d_in[1];
    const int*   row_offsets = (const int*)d_in[2];
    const int*   cols        = (const int*)d_in[3];
    float*       out         = (float*)d_out;
    unsigned short* Wbf      = (unsigned short*)d_ws;  // 33.5 MB of workspace

    scatter_w_kernel<<<M_DIM, 256, 0, stream>>>(values, row_offsets, cols, Wbf);

    dim3 grid(M_DIM / BM, N_DIM / BN, B_DIM);
    spmm_gemm_kernel<<<grid, 256, 0, stream>>>(x, Wbf, out);
}
